// SS2D_8392366097071
// MI455X (gfx1250) — matrix operand order repaired and checked by reading, not compiled, not run on hardware
//
#include <hip/hip_runtime.h>
#include <hip/hip_bf16.h>
#include <math.h>

// ---------------- problem constants ----------------
#define D_MODEL 192
#define D_INNER 384
#define D_STATE 16
#define DT_RANK 12
#define BATCH   8
#define SEQLEN  4096              // 64*64
#define ROWS    (BATCH * SEQLEN)  // 32768
#define XZ_W    (2 * D_INNER)     // 768
#define XDBL_W  48                // dt_rank + 2*d_state = 44, padded to 48

typedef __bf16 v16bf __attribute__((ext_vector_type(16)));
typedef float  v8f   __attribute__((ext_vector_type(8)));
typedef unsigned v4u __attribute__((ext_vector_type(4)));
typedef unsigned v8u __attribute__((ext_vector_type(8)));

union FragBF { v16bf v; unsigned u[8]; v4u q[2]; };

static __device__ __forceinline__ unsigned short f2bf(float f) {
    unsigned u = __float_as_uint(f);
    unsigned r = 0x7FFFu + ((u >> 16) & 1u);
    return (unsigned short)((u + r) >> 16);
}
static __device__ __forceinline__ float silu_f(float x) {
    return x / (1.0f + __expf(-x));
}
static __device__ __forceinline__ float softplus_f(float x) {
    return (x > 20.0f) ? x : log1pf(__expf(x));
}

// ---------------- fp32 -> bf16 convert (with zero pad tail) ----------------
__global__ void cvt_bf16_pad(const float* __restrict__ in,
                             unsigned short* __restrict__ out,
                             int nValid, int nTotal) {
    int i = blockIdx.x * blockDim.x + threadIdx.x;
    if (i < nTotal) out[i] = (i < nValid) ? f2bf(in[i]) : (unsigned short)0;
}

// ---------------- LayerNorm over C=192, reading x as (B,C,H*W) -------------
__global__ void ln_kernel(const float* __restrict__ x,
                          const float* __restrict__ g,
                          const float* __restrict__ b,
                          unsigned short* __restrict__ seq_bf) {
    int row  = blockIdx.x * 8 + (threadIdx.x >> 5);
    int lane = threadIdx.x & 31;
    int bi = row >> 12;
    int l  = row & 4095;
    const float* base = x + (size_t)bi * D_MODEL * SEQLEN + l;
    float v[6];
    float s = 0.0f;
#pragma unroll
    for (int i = 0; i < 6; i++) {
        v[i] = base[(size_t)(lane + 32 * i) * SEQLEN];
        s += v[i];
    }
#pragma unroll
    for (int m = 16; m >= 1; m >>= 1) s += __shfl_xor(s, m, 32);
    float mu = s * (1.0f / D_MODEL);
    float q = 0.0f;
#pragma unroll
    for (int i = 0; i < 6; i++) { float d = v[i] - mu; q += d * d; }
#pragma unroll
    for (int m = 16; m >= 1; m >>= 1) q += __shfl_xor(q, m, 32);
    float rs = rsqrtf(q * (1.0f / D_MODEL) + 1e-5f);
    unsigned short* orow = seq_bf + (size_t)row * D_MODEL;
#pragma unroll
    for (int i = 0; i < 6; i++) {
        int c = lane + 32 * i;
        orow[c] = f2bf((v[i] - mu) * rs * g[c] + b[c]);
    }
}

// ---------------- WMMA GEMM with TDM-staged weight panel -------------------
// Block = 4 waves, each wave owns one 16-row M tile; all waves share one
// weight panel of (TN*16) rows x K cols staged in LDS by the Tensor Data
// Mover (tensor_load_to_lds, D# built in SGPRs), then consumed as WMMA B
// fragments via ds_load_b128 pairs.
// A fragment layout (16-bit A 16x32, same contiguous k order as B): lanes 0-15 row M=lane,
//   v0..7 = K 0..15 (pairs); lanes 16-31 hold K 16..31.
// B fragment: lane n = lane&15 is the output column (= W row); lanes 0-15
//   hold K 0..15 in v0..7, lanes 16-31 hold K 16..31.
template <int TN, int STORE_T>
__global__ void gemm_wmma_tdm(const unsigned short* __restrict__ A,
                              const unsigned short* __restrict__ W,
                              float* __restrict__ C,
                              int M, int K, int Nvalid, int ldc, int nGroups) {
    __shared__ __attribute__((aligned(16))) unsigned short wtile[64 * 384];

    int wave = threadIdx.x >> 5;
    int lane = threadIdx.x & 31;
    int mBlock = blockIdx.x / nGroups;
    int nGroup = blockIdx.x % nGroups;
    int rowsTile = TN * 16;

    // --- stage weight panel into LDS with the Tensor Data Mover (wave 0) ---
    if (wave == 0) {
        unsigned long long gaddr =
            (unsigned long long)(const void*)W +
            (unsigned long long)nGroup * rowsTile * K * 2ull;
        unsigned lds_off = (unsigned)(unsigned long long)(&wtile[0]);
        v4u g0;
        g0[0] = 1u;                                    // count=1, user descriptor
        g0[1] = lds_off;                               // lds_addr (bytes)
        g0[2] = (unsigned)(gaddr & 0xFFFFFFFFull);     // global_addr[31:0]
        g0[3] = (unsigned)((gaddr >> 32) & 0x1FFFFFFull) | (2u << 30); // type=2
        v8u g1;
        g1[0] = (1u << 16);                            // data_size=1 (2 bytes)
        g1[1] = ((unsigned)K & 0xFFFFu) << 16;         // tensor_dim0[15:0]
        g1[2] = (((unsigned)K >> 16) & 0xFFFFu) |
                (((unsigned)rowsTile & 0xFFFFu) << 16);// dim0 hi | tensor_dim1 lo
        g1[3] = (((unsigned)K & 0xFFFFu) << 16);       // tensor_dim1 hi=0 | tile_dim0=K
        g1[4] = ((unsigned)rowsTile & 0xFFFFu);        // tile_dim1 | tile_dim2=0
        g1[5] = (unsigned)K;                           // tensor_dim0_stride[31:0]
        g1[6] = 0u;                                    // stride hi | dim1_stride lo
        g1[7] = 0u;
        asm volatile("tensor_load_to_lds %0, %1" :: "s"(g0), "s"(g1) : "memory");
        __builtin_amdgcn_s_wait_tensorcnt(0);
    }
    __syncthreads();

    int mTile = mBlock * 4 + wave;
    if (mTile * 16 >= M) return;
    int mBase = mTile * 16;
    int half  = lane >> 4;   // K-half selector
    int r     = lane & 15;   // row (A) / column (B/C) within tile

    v8f acc[TN] = {};

    const unsigned short* arow = A + (size_t)(mBase + r) * K;
    for (int k0 = 0; k0 < K; k0 += 32) {
        FragBF a;
#pragma unroll
        for (int i = 0; i < 8; i++) {
            int kk = half * 16 + 2 * i;
            a.u[i] = *(const unsigned*)(arow + k0 + kk);
        }
#pragma unroll
        for (int t = 0; t < TN; t++) {
            // 16 contiguous bf16, 16B-aligned -> two ds_load_b128
            const unsigned short* wrow =
                &wtile[(t * 16 + r) * K + k0 + half * 16];
            FragBF bf;
            bf.q[0] = *(const v4u*)(wrow);
            bf.q[1] = *(const v4u*)(wrow + 8);
            acc[t] = __builtin_amdgcn_wmma_f32_16x16x32_bf16(
                false, a.v, false, bf.v, (short)0, acc[t], false, false);
        }
    }

    // C/D layout: lane n = lane&15, VGPR j holds row M = j + 8*half
#pragma unroll
    for (int t = 0; t < TN; t++) {
        int n = nGroup * (TN * 16) + t * 16 + r;
        if (n >= Nvalid) continue;
        if (STORE_T == 0) {
#pragma unroll
            for (int j = 0; j < 8; j++) {
                int m = mBase + j + half * 8;
                C[(size_t)m * ldc + n] = acc[t][j];
            }
        } else {
            // out_proj: scatter into (b, c, h*w) output layout
#pragma unroll
            for (int j = 0; j < 8; j++) {
                int m  = mBase + j + half * 8;
                int bb = m >> 12;
                int l  = m & 4095;
                C[((size_t)bb * D_MODEL + n) * SEQLEN + l] = acc[t][j];
            }
        }
    }
}

// ---------------- depthwise causal conv (k=4) + bias + SiLU ----------------
__global__ void conv_silu_kernel(const float* __restrict__ xz,
                                 const float* __restrict__ cw,
                                 const float* __restrict__ cb,
                                 float* __restrict__ u,
                                 unsigned short* __restrict__ u_bf) {
    long long idx = (long long)blockIdx.x * blockDim.x + threadIdx.x;
    if (idx >= (long long)ROWS * D_INNER) return;
    int d   = (int)(idx % D_INNER);
    int row = (int)(idx / D_INNER);
    int l   = row & 4095;
    int bL  = row - l;
    float acc = cb[d];
#pragma unroll
    for (int j = 0; j < 4; j++) {
        int l2 = l - 3 + j;
        if (l2 >= 0)
            acc += xz[(size_t)(bL + l2) * XZ_W + d] * cw[d * 4 + j];
    }
    float s = silu_f(acc);
    u[idx]    = s;
    u_bf[idx] = f2bf(s);
}

// ---------------- dt = softplus(x_dbl[:, :12] @ dt_w^T + dt_b) -------------
__global__ void dt_kernel(const float* __restrict__ x_dbl,
                          const float* __restrict__ dtw,
                          const float* __restrict__ dtb,
                          float* __restrict__ dt) {
    long long idx = (long long)blockIdx.x * blockDim.x + threadIdx.x;
    if (idx >= (long long)ROWS * D_INNER) return;
    int d   = (int)(idx % D_INNER);
    int row = (int)(idx / D_INNER);
    const float* xr = x_dbl + (size_t)row * XDBL_W;
    float s = dtb[d];
#pragma unroll
    for (int r = 0; r < DT_RANK; r++) s += xr[r] * dtw[d * DT_RANK + r];
    dt[idx] = softplus_f(s);
}

// ---------------- selective scan: 16 lanes = 16 states of one (b,d) --------
__global__ void scan_kernel(const float* __restrict__ dt,
                            const float* __restrict__ u,
                            const float* __restrict__ x_dbl,
                            const float* __restrict__ A_log,
                            float* __restrict__ ys) {
    int ch = blockIdx.x * (blockDim.x >> 4) + (threadIdx.x >> 4);
    int s  = threadIdx.x & 15;
    int b  = ch / D_INNER;
    int d  = ch % D_INNER;
    float Acoef = -__expf(A_log[d * D_STATE + s]);
    float h = 0.0f;
    const float* dtp = dt + (size_t)b * SEQLEN * D_INNER + d;
    const float* up  = u  + (size_t)b * SEQLEN * D_INNER + d;
    const float* xdp = x_dbl + (size_t)b * SEQLEN * XDBL_W;
    float* yp = ys + (size_t)b * SEQLEN * D_INNER + d;
    for (int l = 0; l < SEQLEN; l++) {
        if (l + 16 < SEQLEN) {
            __builtin_prefetch(dtp + (size_t)(l + 16) * D_INNER, 0, 1);
            __builtin_prefetch(xdp + (size_t)(l + 16) * XDBL_W, 0, 1);
        }
        float dtv = dtp[(size_t)l * D_INNER];
        float uv  = up[(size_t)l * D_INNER];
        float Bv  = xdp[(size_t)l * XDBL_W + DT_RANK + s];
        float Cv  = xdp[(size_t)l * XDBL_W + DT_RANK + D_STATE + s];
        float dA = __expf(dtv * Acoef);
        h = h * dA + dtv * uv * Bv;
        float p = h * Cv;
        p += __shfl_xor(p, 8, 16);
        p += __shfl_xor(p, 4, 16);
        p += __shfl_xor(p, 2, 16);
        p += __shfl_xor(p, 1, 16);
        if (s == 0) yp[(size_t)l * D_INNER] = p;
    }
}

// ---------------- y = (ys + u*Dp) * silu(z) -> bf16 ------------------------
__global__ void gate_kernel(const float* __restrict__ ys,
                            const float* __restrict__ u,
                            const float* __restrict__ Dp,
                            const float* __restrict__ xz,
                            unsigned short* __restrict__ y_bf) {
    long long idx = (long long)blockIdx.x * blockDim.x + threadIdx.x;
    if (idx >= (long long)ROWS * D_INNER) return;
    int d   = (int)(idx % D_INNER);
    int row = (int)(idx / D_INNER);
    float z = xz[(size_t)row * XZ_W + D_INNER + d];
    float y = (ys[idx] + u[idx] * Dp[d]) * silu_f(z);
    y_bf[idx] = f2bf(y);
}

// ---------------- host side ----------------
extern "C" void kernel_launch(void* const* d_in, const int* in_sizes, int n_in,
                              void* d_out, int out_size, void* d_ws, size_t ws_size,
                              hipStream_t stream) {
    (void)in_sizes; (void)n_in; (void)out_size; (void)ws_size;
    const float* x        = (const float*)d_in[0];
    const float* ln_g     = (const float*)d_in[1];
    const float* ln_b     = (const float*)d_in[2];
    const float* in_w     = (const float*)d_in[3];
    const float* conv_w   = (const float*)d_in[4];
    const float* conv_b   = (const float*)d_in[5];
    const float* xp_w     = (const float*)d_in[6];
    const float* dt_w     = (const float*)d_in[7];
    const float* dt_b     = (const float*)d_in[8];
    const float* A_log    = (const float*)d_in[9];
    const float* Dp       = (const float*)d_in[10];
    const float* out_w    = (const float*)d_in[11];
    float* out            = (float*)d_out;

    char* ws = (char*)d_ws;
    size_t off = 0;
    auto alloc = [&](size_t bytes) -> void* {
        void* p = ws + off;
        off = (off + bytes + 255) & ~(size_t)255;
        return p;
    };
    unsigned short* w_in_bf  = (unsigned short*)alloc((size_t)XZ_W * D_MODEL * 2);
    unsigned short* w_xp_bf  = (unsigned short*)alloc((size_t)XDBL_W * D_INNER * 2);
    unsigned short* w_out_bf = (unsigned short*)alloc((size_t)D_MODEL * D_INNER * 2);
    unsigned short* seq_bf   = (unsigned short*)alloc((size_t)ROWS * D_MODEL * 2);
    float*          xz       = (float*)alloc((size_t)ROWS * XZ_W * 4);
    float*          u        = (float*)alloc((size_t)ROWS * D_INNER * 4);
    unsigned short* u_bf     = (unsigned short*)alloc((size_t)ROWS * D_INNER * 2);
    float*          x_dbl    = (float*)alloc((size_t)ROWS * XDBL_W * 4);
    float*          dt       = (float*)alloc((size_t)ROWS * D_INNER * 4);
    float*          ys       = (float*)alloc((size_t)ROWS * D_INNER * 4);
    unsigned short* y_bf     = (unsigned short*)alloc((size_t)ROWS * D_INNER * 2);

    // weight conversions (x_proj padded 44 -> 48 rows with zeros)
    {
        int n = XZ_W * D_MODEL;
        cvt_bf16_pad<<<(n + 255) / 256, 256, 0, stream>>>(in_w, w_in_bf, n, n);
    }
    {
        int nv = (DT_RANK + 2 * D_STATE) * D_INNER;   // 44*384
        int nt = XDBL_W * D_INNER;                    // 48*384
        cvt_bf16_pad<<<(nt + 255) / 256, 256, 0, stream>>>(xp_w, w_xp_bf, nv, nt);
    }
    {
        int n = D_MODEL * D_INNER;
        cvt_bf16_pad<<<(n + 255) / 256, 256, 0, stream>>>(out_w, w_out_bf, n, n);
    }

    // LayerNorm -> bf16 activations
    ln_kernel<<<ROWS / 8, 256, 0, stream>>>(x, ln_g, ln_b, seq_bf);

    // in_proj: [ROWS x 192] * [768 x 192]^T -> xz [ROWS x 768]
    {
        int nGroups = XZ_W / 64;                 // 12
        int blocks  = (ROWS / 64) * nGroups;     // 4 M-tiles per block
        gemm_wmma_tdm<4, 0><<<blocks, 128, 0, stream>>>(
            seq_bf, w_in_bf, xz, ROWS, D_MODEL, XZ_W, XZ_W, nGroups);
    }

    // depthwise conv + SiLU -> u (fp32 + bf16)
    {
        long long n = (long long)ROWS * D_INNER;
        conv_silu_kernel<<<(unsigned)((n + 255) / 256), 256, 0, stream>>>(
            xz, conv_w, conv_b, u, u_bf);
    }

    // x_proj: [ROWS x 384] * [48 x 384]^T -> x_dbl [ROWS x 48] (cols 0..43 valid)
    {
        int nGroups = 1;                         // 48 cols = 3 tiles of 16
        int blocks  = (ROWS / 64) * nGroups;
        gemm_wmma_tdm<3, 0><<<blocks, 128, 0, stream>>>(
            u_bf, w_xp_bf, x_dbl, ROWS, D_INNER, DT_RANK + 2 * D_STATE,
            XDBL_W, nGroups);
    }

    // dt projection + softplus
    {
        long long n = (long long)ROWS * D_INNER;
        dt_kernel<<<(unsigned)((n + 255) / 256), 256, 0, stream>>>(
            x_dbl, dt_w, dt_b, dt);
    }

    // selective scan: 3072 channels, 16 lanes each -> 192 blocks of 256
    scan_kernel<<<(BATCH * D_INNER) / 16, 256, 0, stream>>>(dt, u, x_dbl, A_log, ys);

    // gating -> bf16
    {
        long long n = (long long)ROWS * D_INNER;
        gate_kernel<<<(unsigned)((n + 255) / 256), 256, 0, stream>>>(
            ys, u, Dp, xz, y_bf);
    }

    // out_proj: [ROWS x 384] * [192 x 384]^T -> out, stored transposed (b,c,l)
    {
        int nGroups = D_MODEL / 64;              // 3
        int blocks  = (ROWS / 64) * nGroups;
        gemm_wmma_tdm<4, 1><<<blocks, 128, 0, stream>>>(
            y_bf, w_out_bf, out, ROWS, D_INNER, D_MODEL, D_MODEL, nGroups);
    }
}
